// ODEFunc_35570919146262
// MI455X (gfx1250) — matrix operand order repaired and checked by reading, not compiled, not run on hardware
//
#include <hip/hip_runtime.h>
#include <math.h>
#include <stdint.h>

// ---------------- problem constants ----------------
constexpr int BB = 1024;   // batch
constexpr int TT = 32;     // timesteps
constexpr int XS = 128;    // X_SIZE
constexpr int AS = 128;    // A_SIZE
constexpr int XHC = 512;   // XH
constexpr int AHC = 512;   // AH
constexpr int DD = XHC + AHC + XS;  // 1152
constexpr int G3 = 3 * XHC;         // 1536

typedef _Float16 half_t;
typedef half_t v16h __attribute__((ext_vector_type(16)));
typedef half_t v8h  __attribute__((ext_vector_type(8)));
typedef float  v8f  __attribute__((ext_vector_type(8)));

#define LDS_STRIDE 56              // halves: 112B pitch, 16B-aligned, conflict-free
#define STAGE_HALVES (256 * LDS_STRIDE)   // A(128 rows) + B(128 rows) per buffer
#define STAGE_BYTES  (STAGE_HALVES * 2)   // 28672 B

// async 16B global -> LDS copy (CDNA5 GLOBAL_LOAD_ASYNC_TO_LDS_B128, ASYNCcnt)
__device__ __forceinline__ void async_cp16(uint32_t lds_off, const half_t* gsrc) {
  asm volatile("global_load_async_to_lds_b128 %0, %1, off"
               :: "v"(lds_off), "v"(gsrc) : "memory");
}

__device__ __forceinline__ void wait_async0() {
#if __has_builtin(__builtin_amdgcn_s_wait_asynccnt)
  __builtin_amdgcn_s_wait_asynccnt(0);
#else
  asm volatile("s_wait_asynccnt 0" ::: "memory");
#endif
}

// LDS aperture keeps the allocation-relative byte offset in addr[31:0]
__device__ __forceinline__ uint32_t lds_off32(const void* p) {
  return (uint32_t)(uintptr_t)p;
}

// ---------------------------------------------------------------------------
// WMMA GEMM:  C[M x N] = A[M x K] * W[N x K]^T (+bias), f16 in, f32 acc.
// Block = 256 threads (8 waves), tile 128x128; wave tile 32x64 (8 accs).
// Double-buffered LDS fed by async load-to-LDS, one barrier per K-chunk.
// mode 0: Cf = acc + bias (f32); mode 1: Ch = f16(softplus(acc + bias)).
// ---------------------------------------------------------------------------
__global__ __launch_bounds__(256) void gemm_wmma_f16(
    const half_t* __restrict__ A, int lda,
    const half_t* __restrict__ W, int ldb,
    const float* __restrict__ bias,
    float* __restrict__ Cf, half_t* __restrict__ Ch, int ldc,
    int K, int mode)
{
  __shared__ alignas(16) half_t smem[2][STAGE_HALVES];

  const int tid  = threadIdx.x;
  const int lane = tid & 31;
  const int wid  = tid >> 5;
  const int wm   = (wid >> 1) * 32;   // wave row offset in 128-row tile
  const int wn   = (wid & 1) * 64;    // wave col offset in 128-col tile
  const int mb   = blockIdx.y * 128;
  const int nb   = blockIdx.x * 128;
  const int r    = lane & 15;
  const int hh   = lane >> 4;

  // per-thread copy assignment: 4 async B128 per stage
  const int crow = tid >> 2;          // 0..63
  const int ckc  = (tid & 3) * 8;     // 0,8,16,24 halves
  const half_t* gA0 = A + (size_t)(mb + crow) * lda + ckc;
  const half_t* gA1 = A + (size_t)(mb + crow + 64) * lda + ckc;
  const half_t* gB0 = W + (size_t)(nb + crow) * ldb + ckc;
  const half_t* gB1 = W + (size_t)(nb + crow + 64) * ldb + ckc;
  const uint32_t lA0 = lds_off32(&smem[0][crow * LDS_STRIDE + ckc]);
  const uint32_t lA1 = lds_off32(&smem[0][(crow + 64) * LDS_STRIDE + ckc]);
  const uint32_t lB0 = lds_off32(&smem[0][(128 + crow) * LDS_STRIDE + ckc]);
  const uint32_t lB1 = lds_off32(&smem[0][(128 + crow + 64) * LDS_STRIDE + ckc]);

  v8f acc[2][4] = {};

  const int nstage = K >> 5;  // K / 32
  // prologue: stage 0 into buffer 0
  {
    async_cp16(lA0, gA0);
    async_cp16(lA1, gA1);
    async_cp16(lB0, gB0);
    async_cp16(lB1, gB1);
  }

  int buf = 0;
  for (int it = 0; it < nstage; ++it) {
    wait_async0();       // my async writes into buf are in LDS
    __syncthreads();     // everyone's writes visible; prev compute done
    if (it + 1 < nstage) {
      const int kn = (it + 1) << 5;
      const uint32_t bo = (buf ^ 1) ? (uint32_t)STAGE_BYTES : 0u;
      async_cp16(lA0 + bo, gA0 + kn);
      async_cp16(lA1 + bo, gA1 + kn);
      async_cp16(lB0 + bo, gB0 + kn);
      async_cp16(lB1 + bo, gB1 + kn);
    }

    const half_t* sa = &smem[buf][0];
    const half_t* sb = sa + 128 * LDS_STRIDE;

    // A frags: lane half hh holds K = 8*hh..+7 and 16+8*hh..+7
    v16h af[2];
    #pragma unroll
    for (int tm = 0; tm < 2; ++tm) {
      const int row = wm + tm * 16 + r;
      v8h lo = *(const v8h*)&sa[row * LDS_STRIDE + hh * 8];
      v8h hi = *(const v8h*)&sa[row * LDS_STRIDE + 16 + hh * 8];
      af[tm] = __builtin_shufflevector(lo, hi, 0, 1, 2, 3, 4, 5, 6, 7,
                                       8, 9, 10, 11, 12, 13, 14, 15);
    }
    // B frags: lane half hh holds K = 8*hh..+7 and 16+8*hh..+7
    v16h bf[4];
    #pragma unroll
    for (int tn = 0; tn < 4; ++tn) {
      const int col = wn + tn * 16 + r;
      v8h blo = *(const v8h*)&sb[col * LDS_STRIDE + hh * 8];
      v8h bhi = *(const v8h*)&sb[col * LDS_STRIDE + 16 + hh * 8];
      bf[tn] = __builtin_shufflevector(blo, bhi, 0, 1, 2, 3, 4, 5, 6, 7,
                                       8, 9, 10, 11, 12, 13, 14, 15);
    }
    #pragma unroll
    for (int tm = 0; tm < 2; ++tm)
      #pragma unroll
      for (int tn = 0; tn < 4; ++tn)
        acc[tm][tn] = __builtin_amdgcn_wmma_f32_16x16x32_f16(
            false, af[tm], false, bf[tn], (short)0, acc[tm][tn], false, false);

    buf ^= 1;
  }

  // epilogue: acc VGPR e holds row (e + 8*hh), col = lane&15
  #pragma unroll
  for (int tm = 0; tm < 2; ++tm) {
    #pragma unroll
    for (int tn = 0; tn < 4; ++tn) {
      const int col     = nb + wn + tn * 16 + r;
      const int rowbase = mb + wm + tm * 16 + hh * 8;
      const float bv    = bias[col];
      #pragma unroll
      for (int e = 0; e < 8; ++e) {
        const float v   = acc[tm][tn][e] + bv;
        const size_t ix = (size_t)(rowbase + e) * ldc + col;
        if (mode == 0) {
          Cf[ix] = v;
        } else {
          const float sp = (v > 20.f) ? v : log1pf(expf(v));
          Ch[ix] = (half_t)sp;
        }
      }
    }
  }
}

// ---------------- elementwise kernels ----------------
__global__ void k_f32_to_f16(const float* __restrict__ src,
                             half_t* __restrict__ dst, int n) {
  int i = blockIdx.x * blockDim.x + threadIdx.x;
  if (i < n) dst[i] = (half_t)src[i];
}

// init s = [0, 0, x[:,0,:]]; write x_list[:,0,:] and z_a_seq[:,0,:] = 0
__global__ void k_init(const float* __restrict__ x, float* __restrict__ s,
                       half_t* __restrict__ sh, float* __restrict__ xout,
                       float* __restrict__ zaout) {
  int i = blockIdx.x * blockDim.x + threadIdx.x;
  if (i >= BB * DD) return;
  int b = i / DD, col = i % DD;
  float v = 0.f;
  if (col >= XHC + AHC) {
    v = x[(size_t)b * TT * XS + (col - XHC - AHC)];
    xout[(size_t)b * TT * XS + (col - XHC - AHC)] = v;
  } else if (col >= XHC) {
    zaout[(size_t)b * TT * AHC + (col - XHC)] = 0.f;
  }
  s[i] = v;
  sh[i] = (half_t)v;
}

// mask[b] = (sum_k y[b,t,k] != 0)
__global__ void k_mask(const float* __restrict__ y, float* __restrict__ m, int t) {
  int b = blockIdx.x * blockDim.x + threadIdx.x;
  if (b >= BB) return;
  const float* p = y + (size_t)b * TT * AS + (size_t)t * AS;
  float sm = 0.f;
  for (int k = 0; k < AS; ++k) sm += p[k];
  m[b] = (sm != 0.f) ? 1.f : 0.f;
}

// cat[b, 0:128] = xt (from s f32), cat[b,128:256] = y[b,t,:]  (f16)
__global__ void k_cat(const float* __restrict__ s, const float* __restrict__ y,
                      half_t* __restrict__ cat, int t) {
  int i = blockIdx.x * blockDim.x + threadIdx.x;
  if (i >= BB * (XS + AS)) return;
  int b = i / (XS + AS), j = i % (XS + AS);
  float v = (j < XS) ? s[(size_t)b * DD + XHC + AHC + j]
                     : y[(size_t)b * TT * AS + (size_t)t * AS + (j - XS)];
  cat[i] = (half_t)v;
}

// GRU gate combine; writes h_new into s[:, hoff:hoff+512] (f32 + f16).
__global__ void k_gru(const float* __restrict__ gi, const float* __restrict__ gh,
                      float* __restrict__ s, half_t* __restrict__ sh, int hoff,
                      const float* __restrict__ mask, float* __restrict__ zout) {
  int i = blockIdx.x * blockDim.x + threadIdx.x;
  if (i >= BB * XHC) return;
  int b = i / XHC, j = i % XHC;
  size_t gb = (size_t)b * G3;
  float ir = gi[gb + j], iz = gi[gb + XHC + j], inn = gi[gb + 2 * XHC + j];
  float hr = gh[gb + j], hz = gh[gb + XHC + j], hn = gh[gb + 2 * XHC + j];
  float rr = 1.f / (1.f + expf(-(ir + hr)));
  float zz = 1.f / (1.f + expf(-(iz + hz)));
  float nn = tanhf(inn + rr * hn);
  size_t si = (size_t)b * DD + hoff + j;
  float hold = s[si];
  float hnew = (1.f - zz) * nn + zz * hold;
  if (mask) hnew = (mask[b] != 0.f) ? hnew : hold;
  s[si] = hnew;
  sh[si] = (half_t)hnew;
  if (zout) zout[(size_t)b * TT * AHC + j] = hnew;
}

// s_eval = f16( s + h*(a1*k1 + a2*k2 + a3*k3) ),  h = ts[t+1]-ts[t]
__global__ void k_evalin(const float* __restrict__ s, const float* __restrict__ k1,
                         const float* __restrict__ k2, const float* __restrict__ k3,
                         const float* __restrict__ ts, int t,
                         float a1, float a2, float a3, half_t* __restrict__ sev) {
  int i = blockIdx.x * blockDim.x + threadIdx.x;
  if (i >= BB * DD) return;
  float h = ts[t + 1] - ts[t];
  float acc = a1 * k1[i];
  if (k2) acc += a2 * k2[i];
  if (k3) acc += a3 * k3[i];
  sev[i] = (half_t)(s[i] + h * acc);
}

// s = s + h/8*(k1 + 3*(k2+k3) + k4); scatter x slice into x_list[:, t+1, :]
__global__ void k_final(float* __restrict__ s, half_t* __restrict__ sh,
                        const float* __restrict__ k1, const float* __restrict__ k2,
                        const float* __restrict__ k3, const float* __restrict__ k4,
                        const float* __restrict__ ts, int t, float* __restrict__ xout) {
  int i = blockIdx.x * blockDim.x + threadIdx.x;
  if (i >= BB * DD) return;
  float h = ts[t + 1] - ts[t];
  float v = s[i] + (h / 8.f) * (k1[i] + 3.f * (k2[i] + k3[i]) + k4[i]);
  s[i] = v;
  sh[i] = (half_t)v;
  int col = i % DD;
  if (col >= XHC + AHC) {
    int b = i / DD;
    xout[(size_t)b * TT * XS + (col - XHC - AHC)] = v;
  }
}

// ---------------------------------------------------------------------------
extern "C" void kernel_launch(void* const* d_in, const int* in_sizes, int n_in,
                              void* d_out, int out_size, void* d_ws, size_t ws_size,
                              hipStream_t stream) {
  (void)in_sizes; (void)n_in; (void)out_size; (void)ws_size;
  const float* x     = (const float*)d_in[0];
  const float* y     = (const float*)d_in[1];
  const float* ts    = (const float*)d_in[2];
  const float* Wiha  = (const float*)d_in[3];
  const float* Whha  = (const float*)d_in[4];
  const float* biha  = (const float*)d_in[5];
  const float* bhha  = (const float*)d_in[6];
  const float* Wihx  = (const float*)d_in[7];
  const float* Whhx  = (const float*)d_in[8];
  const float* bihx  = (const float*)d_in[9];
  const float* bhhx  = (const float*)d_in[10];
  const float* W1    = (const float*)d_in[11];
  const float* b1    = (const float*)d_in[12];
  const float* W2    = (const float*)d_in[13];
  const float* b2    = (const float*)d_in[14];

  float* xout  = (float*)d_out;                       // (B, T, XS)
  float* zaout = xout + (size_t)BB * TT * XS;         // (B, T, AH)

  char* ws = (char*)d_ws;
  size_t off = 0;
  auto alloc = [&](size_t bytes) -> void* {
    void* p = ws + off;
    off = (off + bytes + 255) & ~(size_t)255;
    return p;
  };

  half_t* wWiha = (half_t*)alloc((size_t)G3 * (XS + AS) * 2);
  half_t* wWhha = (half_t*)alloc((size_t)G3 * AHC * 2);
  half_t* wWihx = (half_t*)alloc((size_t)G3 * XS * 2);
  half_t* wWhhx = (half_t*)alloc((size_t)G3 * XHC * 2);
  half_t* wW1   = (half_t*)alloc((size_t)DD * DD * 2);
  half_t* wW2   = (half_t*)alloc((size_t)DD * DD * 2);
  float*  s32   = (float*)alloc((size_t)BB * DD * 4);
  half_t* s16   = (half_t*)alloc((size_t)BB * DD * 2);
  half_t* cat   = (half_t*)alloc((size_t)BB * (XS + AS) * 2);
  float*  gi    = (float*)alloc((size_t)BB * G3 * 4);
  float*  gh    = (float*)alloc((size_t)BB * G3 * 4);
  half_t* ubuf  = (half_t*)alloc((size_t)BB * DD * 2);
  half_t* sev   = (half_t*)alloc((size_t)BB * DD * 2);
  float*  kb1   = (float*)alloc((size_t)BB * DD * 4);
  float*  kb2   = (float*)alloc((size_t)BB * DD * 4);
  float*  kb3   = (float*)alloc((size_t)BB * DD * 4);
  float*  kb4   = (float*)alloc((size_t)BB * DD * 4);
  float*  maskb = (float*)alloc((size_t)BB * 4);

  auto cvt = [&](const float* src, half_t* dst, int n) {
    k_f32_to_f16<<<(n + 255) / 256, 256, 0, stream>>>(src, dst, n);
  };
  cvt(Wiha, wWiha, G3 * (XS + AS));
  cvt(Whha, wWhha, G3 * AHC);
  cvt(Wihx, wWihx, G3 * XS);
  cvt(Whhx, wWhhx, G3 * XHC);
  cvt(W1, wW1, DD * DD);
  cvt(W2, wW2, DD * DD);

  k_init<<<(BB * DD + 255) / 256, 256, 0, stream>>>(x, s32, s16, xout, zaout);

  auto gemm = [&](const half_t* A, int lda, const half_t* W, int ldb,
                  const float* bias, float* Cf, half_t* Ch, int ldc,
                  int M, int N, int K, int mode) {
    dim3 grid(N / 128, M / 128), block(256);
    gemm_wmma_f16<<<grid, block, 0, stream>>>(A, lda, W, ldb, bias, Cf, Ch,
                                              ldc, K, mode);
  };

  const int EW = (BB * DD + 255) / 256;
  const int GW = (BB * XHC + 255) / 256;

  for (int t = 0; t < TT - 1; ++t) {
    // ---- GRU A: input = concat(xt, a_t), hidden = z_a (s cols 512:1024) ----
    k_mask<<<(BB + 255) / 256, 256, 0, stream>>>(y, maskb, t);
    k_cat<<<(BB * (XS + AS) + 255) / 256, 256, 0, stream>>>(s32, y, cat, t);
    gemm(cat, XS + AS, wWiha, XS + AS, biha, gi, nullptr, G3, BB, G3, XS + AS, 0);
    gemm(s16 + XHC, DD, wWhha, AHC, bhha, gh, nullptr, G3, BB, G3, AHC, 0);
    k_gru<<<GW, 256, 0, stream>>>(gi, gh, s32, s16, XHC, maskb,
                                  zaout + (size_t)(t + 1) * AHC);
    // ---- GRU X: input = xt (cols 1024:1152), hidden = z_x (cols 0:512) ----
    gemm(s16 + XHC + AHC, DD, wWihx, XS, bihx, gi, nullptr, G3, BB, G3, XS, 0);
    gemm(s16, DD, wWhhx, XHC, bhhx, gh, nullptr, G3, BB, G3, XHC, 0);
    k_gru<<<GW, 256, 0, stream>>>(gi, gh, s32, s16, 0, nullptr, nullptr);

    // ---- RK4 (3/8): f(s) = softplus(s W1^T + b1) W2^T + b2 ----
    gemm(s16, DD, wW1, DD, b1, nullptr, ubuf, DD, BB, DD, DD, 1);
    gemm(ubuf, DD, wW2, DD, b2, kb1, nullptr, DD, BB, DD, DD, 0);

    k_evalin<<<EW, 256, 0, stream>>>(s32, kb1, nullptr, nullptr, ts, t,
                                     1.f / 3.f, 0.f, 0.f, sev);
    gemm(sev, DD, wW1, DD, b1, nullptr, ubuf, DD, BB, DD, DD, 1);
    gemm(ubuf, DD, wW2, DD, b2, kb2, nullptr, DD, BB, DD, DD, 0);

    k_evalin<<<EW, 256, 0, stream>>>(s32, kb1, kb2, nullptr, ts, t,
                                     -1.f / 3.f, 1.f, 0.f, sev);
    gemm(sev, DD, wW1, DD, b1, nullptr, ubuf, DD, BB, DD, DD, 1);
    gemm(ubuf, DD, wW2, DD, b2, kb3, nullptr, DD, BB, DD, DD, 0);

    k_evalin<<<EW, 256, 0, stream>>>(s32, kb1, kb2, kb3, ts, t,
                                     1.f, -1.f, 1.f, sev);
    gemm(sev, DD, wW1, DD, b1, nullptr, ubuf, DD, BB, DD, DD, 1);
    gemm(ubuf, DD, wW2, DD, b2, kb4, nullptr, DD, BB, DD, DD, 0);

    k_final<<<EW, 256, 0, stream>>>(s32, s16, kb1, kb2, kb3, kb4, ts, t,
                                    xout + (size_t)(t + 1) * XS);
  }
}
